// Attention_67018669687187
// MI455X (gfx1250) — matrix operand order repaired and checked by reading, not compiled, not run on hardware
//
#include <hip/hip_runtime.h>

#define B_      16
#define C_      512
#define N_      1024
#define HEADS_  4
#define HD_     128
#define SCALE_  0.08838834764831845f  // 1/sqrt(128)

typedef __attribute__((ext_vector_type(16))) __bf16        v16bf;
typedef __attribute__((ext_vector_type(8)))  float         v8f;
typedef __attribute__((ext_vector_type(4)))  unsigned int  u32x4;

union FragU {
  v16bf          v;
  u32x4          q[2];
  unsigned short h[16];
  unsigned int   w[8];
};

__device__ __forceinline__ unsigned short f2bf(float f) {
  unsigned int x = __float_as_uint(f);
  unsigned int r = x + 0x7FFFu + ((x >> 16) & 1u);   // round-to-nearest-even
  return (unsigned short)(r >> 16);
}

__device__ __forceinline__ v8f wmma_bf16(v16bf a, v16bf b, v8f c) {
  // emits v_wmma_f32_16x16x32_bf16
  return __builtin_amdgcn_wmma_f32_16x16x32_bf16(false, a, false, b, (short)0, c,
                                                 false, false);
}

// A-operand (16x32 bf16): lane l -> row (l&15); 16 contiguous K at (l>>4)*16.
__device__ __forceinline__ v16bf load_fragA(const unsigned short* p, int stride) {
  const int l = threadIdx.x & 31;
  const unsigned short* q = p + (size_t)(l & 15) * stride + ((l >> 4) << 4);
  FragU u;
  u.q[0] = *(const u32x4*)(q);
  u.q[1] = *(const u32x4*)(q + 8);
  return u.v;
}

// B-operand (32x16 bf16): lane l -> col (l&15); 16 contiguous K at (l>>4)*16.
__device__ __forceinline__ v16bf load_fragB(const unsigned short* p, int stride) {
  const int l = threadIdx.x & 31;
  const unsigned short* q = p + (size_t)(l & 15) * stride + ((l >> 4) << 4);
  FragU u;
  u.q[0] = *(const u32x4*)(q);
  u.q[1] = *(const u32x4*)(q + 8);
  return u.v;
}

__device__ __forceinline__ u32x4 pack8(v8f a) {
  u32x4 r;
  r.x = (unsigned)f2bf(a[0]) | ((unsigned)f2bf(a[1]) << 16);
  r.y = (unsigned)f2bf(a[2]) | ((unsigned)f2bf(a[3]) << 16);
  r.z = (unsigned)f2bf(a[4]) | ((unsigned)f2bf(a[5]) << 16);
  r.w = (unsigned)f2bf(a[6]) | ((unsigned)f2bf(a[7]) << 16);
  return r;
}

// ---------------------------------------------------------------- weights->bf16
__global__ void __launch_bounds__(256)
k_convert(const float* __restrict__ qkv_w, const float* __restrict__ proj_w,
          unsigned short* __restrict__ wq_bf, unsigned short* __restrict__ wp_bf) {
  const int i = blockIdx.x * 256 + threadIdx.x;     // 1,048,576 total
  if (i < 3 * C_ * C_) wq_bf[i] = f2bf(qkv_w[i]);
  else                 wp_bf[i - 3 * C_ * C_] = f2bf(proj_w[i - 3 * C_ * C_]);
}

// ------------------------------------------------- GroupNorm -> h_t[b][n][c] bf16
__global__ void __launch_bounds__(256)
k_gn(const float* __restrict__ x, const float* __restrict__ gw,
     const float* __restrict__ gb, unsigned short* __restrict__ h_t) {
  __shared__ float ssum[256];
  __shared__ float ssq[256];
  const int bi = blockIdx.x >> 5;       // batch
  const int g  = blockIdx.x & 31;       // group (16 channels)
  const int tid = threadIdx.x;
  const float* xg = x + ((size_t)(bi * C_ + g * 16)) * N_;

  float s = 0.f, sq = 0.f;
  for (int i = tid; i < 16 * N_; i += 256) {
    float v = xg[i];
    s += v; sq += v * v;
  }
  ssum[tid] = s; ssq[tid] = sq;
  __syncthreads();
  for (int off = 128; off; off >>= 1) {
    if (tid < off) { ssum[tid] += ssum[tid + off]; ssq[tid] += ssq[tid + off]; }
    __syncthreads();
  }
  const float mean = ssum[0] * (1.f / 16384.f);
  const float var  = ssq[0] * (1.f / 16384.f) - mean * mean;
  const float rstd = rsqrtf(var + 1e-5f);

  for (int i = tid; i < 16 * N_; i += 256) {
    const int cl = i >> 10, n = i & (N_ - 1);
    const int c = g * 16 + cl;
    const float v = (xg[i] - mean) * rstd * gw[c] + gb[c];
    h_t[((size_t)(bi * N_ + n)) * C_ + c] = f2bf(v);
  }
}

// ------------------------------------- QKV GEMM: [1536 x 1024] = W x h_t^T per b
// Each wave: 16(o) x 128(n) tile -> 8 independent accumulators per k-step.
__global__ void __launch_bounds__(256)
k_qkv(const unsigned short* __restrict__ h_t, const unsigned short* __restrict__ wq,
      const float* __restrict__ qkv_b, unsigned short* __restrict__ q_t,
      unsigned short* __restrict__ k_t, unsigned short* __restrict__ v_) {
  const int wid = threadIdx.x >> 5;
  const int wg  = blockIdx.x * 8 + wid;        // 12288 waves
  const int b   = wg / (96 * 8);
  const int rem = wg % (96 * 8);
  const int ot  = rem >> 3;                    // 96 o-tiles of 16
  const int ng  = rem & 7;                     // 8 n-groups of 128
  const int obase = ot * 16;
  const int nbase = ng * 128;

  const unsigned short* ap = wq + (size_t)obase * C_;
  const unsigned short* bp = h_t + ((size_t)(b * N_ + nbase)) * C_;

  v8f acc[8] = {};
  for (int k = 0; k < C_; k += 32) {
    __builtin_prefetch(ap + k + 32, 0, 3);     // near-cache prefetch
    v16bf a = load_fragA(ap + k, C_);
#pragma unroll
    for (int s = 0; s < 8; ++s) {
      v16bf bb = load_fragB(bp + (size_t)s * 16 * C_ + k, C_);
      acc[s] = wmma_bf16(a, bb, acc[s]);
    }
  }

  const int ln = threadIdx.x & 15;
  const int half = (threadIdx.x >> 4) & 1;
  float bias[8];
#pragma unroll
  for (int r = 0; r < 8; ++r) bias[r] = qkv_b[obase + half * 8 + r];
#pragma unroll
  for (int s = 0; s < 8; ++s)
#pragma unroll
    for (int r = 0; r < 8; ++r) acc[s][r] += bias[r];

  if (obase < C_) {                             // Q -> q_t[b][h][n][c]
    const int head = obase >> 7, coff = (obase & 127) + half * 8;
#pragma unroll
    for (int s = 0; s < 8; ++s) {
      const int n = nbase + s * 16 + ln;
      *(u32x4*)(q_t + ((size_t)((b * HEADS_ + head) * N_ + n)) * HD_ + coff) =
          pack8(acc[s]);
    }
  } else if (obase < 2 * C_) {                  // K -> k_t[b][h][m][c]
    const int o2 = obase - C_;
    const int head = o2 >> 7, coff = (o2 & 127) + half * 8;
#pragma unroll
    for (int s = 0; s < 8; ++s) {
      const int n = nbase + s * 16 + ln;
      *(u32x4*)(k_t + ((size_t)((b * HEADS_ + head) * N_ + n)) * HD_ + coff) =
          pack8(acc[s]);
    }
  } else {                                      // V -> v[b][h][c][m]
    const int o2 = obase - 2 * C_;
    const int head = o2 >> 7, cb = (o2 & 127) + half * 8;
#pragma unroll
    for (int s = 0; s < 8; ++s) {
      const int n = nbase + s * 16 + ln;
#pragma unroll
      for (int r = 0; r < 8; ++r)
        v_[((size_t)((b * HEADS_ + head) * HD_ + cb + r)) * N_ + n] =
            f2bf(acc[s][r]);
    }
  }
}

// ------------------------------------- flash attention, S^T formulation, per-wave
__global__ void __launch_bounds__(256)
k_attn(const unsigned short* __restrict__ q_t, const unsigned short* __restrict__ k_t,
       const unsigned short* __restrict__ v_, unsigned short* __restrict__ o_nc) {
  const int lane = threadIdx.x & 31;
  const int wid  = threadIdx.x >> 5;
  const int wg   = blockIdx.x * 8 + wid;        // 4096 waves
  const int bh   = wg >> 6;                     // b*4+head
  const int nt   = wg & 63;                     // 64 query tiles of 16
  const int nbase = nt * 16;
  const bool hiHalf = lane >= 16;

  const unsigned short* qp = q_t + ((size_t)bh * N_ + nbase) * HD_;
  v16bf qf[4];
#pragma unroll
  for (int kk = 0; kk < 4; ++kk) qf[kk] = load_fragB(qp + kk * 32, HD_);

  v8f acc[8] = {};
  float runmax = -1e30f, runsum = 0.f;

  for (int mb = 0; mb < N_; mb += 32) {
    const unsigned short* kp = k_t + ((size_t)bh * N_ + mb) * HD_;
    __builtin_prefetch(kp + 32 * HD_, 0, 3);

    // S^T tiles: rows m, cols n (lane = col n, regs = m).
    // Two partial accumulators per tile halve the WMMA RAW chain.
    v8f s0a = {}, s0b = {}, s1a = {}, s1b = {};
    s0a = wmma_bf16(load_fragA(kp + 0 * 32, HD_), qf[0], s0a);
    s0b = wmma_bf16(load_fragA(kp + 1 * 32, HD_), qf[1], s0b);
    s1a = wmma_bf16(load_fragA(kp + 16 * HD_ + 0 * 32, HD_), qf[0], s1a);
    s1b = wmma_bf16(load_fragA(kp + 16 * HD_ + 1 * 32, HD_), qf[1], s1b);
    s0a = wmma_bf16(load_fragA(kp + 2 * 32, HD_), qf[2], s0a);
    s0b = wmma_bf16(load_fragA(kp + 3 * 32, HD_), qf[3], s0b);
    s1a = wmma_bf16(load_fragA(kp + 16 * HD_ + 2 * 32, HD_), qf[2], s1a);
    s1b = wmma_bf16(load_fragA(kp + 16 * HD_ + 3 * 32, HD_), qf[3], s1b);
    v8f s0 = s0a + s0b;
    v8f s1 = s1a + s1b;

    // online softmax over the 32 m-rows of this step (per column n)
    float lm = -1e30f;
#pragma unroll
    for (int r = 0; r < 8; ++r) {
      s0[r] *= SCALE_; s1[r] *= SCALE_;
      lm = fmaxf(lm, fmaxf(s0[r], s1[r]));
    }
    lm = fmaxf(lm, __shfl_xor(lm, 16, 32));
    const float nm = fmaxf(runmax, lm);
    const float fr = __expf(runmax - nm);
    float p0[8], p1[8], ls = 0.f;
#pragma unroll
    for (int r = 0; r < 8; ++r) {
      p0[r] = __expf(s0[r] - nm);
      p1[r] = __expf(s1[r] - nm);
      ls += p0[r] + p1[r];
    }
    ls += __shfl_xor(ls, 16, 32);
    runsum = runsum * fr + ls;
    runmax = nm;
#pragma unroll
    for (int ct = 0; ct < 8; ++ct)
#pragma unroll
      for (int r = 0; r < 8; ++r) acc[ct][r] *= fr;

    // build P^T as a B-operand: lane = col n, K = 32 m-rows.
    // lanes<16 need m0..15 (own m0..7 + partner's m8..15), lanes>=16 need m16..31.
    FragU pf;
#pragma unroll
    for (int i = 0; i < 8; ++i) {
      const float o0 = __shfl_xor(p0[i], 16, 32);
      const float o1 = __shfl_xor(p1[i], 16, 32);
      pf.h[i]     = f2bf(hiHalf ? o1 : p0[i]);
      pf.h[8 + i] = f2bf(hiHalf ? p1[i] : o0);
    }

    // O^T += V(c x m) * P^T(m x n): 8 independent c-tiles cover hd=128
    const unsigned short* vp = v_ + (size_t)bh * HD_ * N_ + mb;
#pragma unroll
    for (int ct = 0; ct < 8; ++ct)
      acc[ct] = wmma_bf16(load_fragA(vp + (size_t)ct * 16 * N_, N_), pf.v, acc[ct]);
  }

  const float inv = 1.f / runsum;
  const int ln = lane & 15, half = lane >> 4;
  const int b = bh >> 2, head = bh & 3;
  unsigned short* ob =
      o_nc + ((size_t)(b * N_ + nbase + ln)) * C_ + head * HD_ + half * 8;
#pragma unroll
  for (int ct = 0; ct < 8; ++ct) {
    v8f t = acc[ct];
#pragma unroll
    for (int r = 0; r < 8; ++r) t[r] *= inv;
    *(u32x4*)(ob + ct * 16) = pack8(t);
  }
}

// ------------------------------------------------ proj GEMM + bias + residual
// Each wave: 16(o) x 128(n) tile.
__global__ void __launch_bounds__(256)
k_proj(const unsigned short* __restrict__ o_nc, const unsigned short* __restrict__ wp,
       const float* __restrict__ proj_b, const float* __restrict__ x,
       float* __restrict__ out) {
  const int wid = threadIdx.x >> 5;
  const int wg  = blockIdx.x * 8 + wid;         // 4096 waves
  const int b   = wg / (32 * 8);
  const int rem = wg % (32 * 8);
  const int ot  = rem >> 3;                     // 32 o-tiles of 16
  const int ng  = rem & 7;                      // 8 n-groups of 128
  const int obase = ot * 16;
  const int nbase = ng * 128;

  const unsigned short* ap = wp + (size_t)obase * C_;
  const unsigned short* bp = o_nc + ((size_t)(b * N_ + nbase)) * C_;

  v8f acc[8] = {};
  for (int k = 0; k < C_; k += 32) {
    __builtin_prefetch(ap + k + 32, 0, 3);
    v16bf a = load_fragA(ap + k, C_);
#pragma unroll
    for (int s = 0; s < 8; ++s) {
      v16bf bb = load_fragB(bp + (size_t)s * 16 * C_ + k, C_);
      acc[s] = wmma_bf16(a, bb, acc[s]);
    }
  }

  const int ln = threadIdx.x & 15;
  const int half = (threadIdx.x >> 4) & 1;
#pragma unroll
  for (int s = 0; s < 8; ++s) {
    const int n = nbase + s * 16 + ln;
#pragma unroll
    for (int r = 0; r < 8; ++r) {
      const int o = obase + half * 8 + r;
      const size_t idx = ((size_t)(b * C_ + o)) * N_ + n;
      out[idx] = x[idx] + proj_b[o] + acc[s][r];
    }
  }
}

extern "C" void kernel_launch(void* const* d_in, const int* in_sizes, int n_in,
                              void* d_out, int out_size, void* d_ws, size_t ws_size,
                              hipStream_t stream) {
  const float* x      = (const float*)d_in[0];
  const float* gn_w   = (const float*)d_in[1];
  const float* gn_b   = (const float*)d_in[2];
  const float* qkv_w  = (const float*)d_in[3];
  const float* qkv_b  = (const float*)d_in[4];
  const float* proj_w = (const float*)d_in[5];
  const float* proj_b = (const float*)d_in[6];
  float* out = (float*)d_out;

  unsigned short* ws = (unsigned short*)d_ws;
  // bf16-element offsets (all 256B aligned)
  unsigned short* h_t   = ws;                  // 8,388,608  (reused as o_nc later)
  unsigned short* wq_bf = ws + 8388608;        //   786,432
  unsigned short* wp_bf = ws + 9175040;        //   262,144
  unsigned short* q_t   = ws + 9437184;        // 8,388,608
  unsigned short* k_t   = ws + 17825792;       // 8,388,608
  unsigned short* v_b   = ws + 26214400;       // 8,388,608  -> total ~69 MB

  k_convert<<<4096, 256, 0, stream>>>(qkv_w, proj_w, wq_bf, wp_bf);
  k_gn<<<512, 256, 0, stream>>>(x, gn_w, gn_b, h_t);
  k_qkv<<<1536, 256, 0, stream>>>(h_t, wq_bf, qkv_b, q_t, k_t, v_b);
  k_attn<<<512, 256, 0, stream>>>(q_t, k_t, v_b, h_t /* o_nc reuses h_t */);
  k_proj<<<512, 256, 0, stream>>>(h_t, wp_bf, proj_b, x, out);
}
